// IHGNN_62989990363242
// MI455X (gfx1250) — matrix operand order repaired and checked by reading, compiled, not run on hardware
//
#include <hip/hip_runtime.h>
#include <hip/hip_bf16.h>

typedef __attribute__((ext_vector_type(16))) _Float16 v16h;
typedef __attribute__((ext_vector_type(8)))  _Float16 v8h;
typedef __attribute__((ext_vector_type(8)))  float    v8f;
typedef __attribute__((ext_vector_type(4)))  float    f32x4;

#define G_       512
#define NPG_     1024
#define KTOP     30
#define DEG_     16
#define EPG      (NPG_ * DEG_)     // 16384 edges per graph
#define FIN      128
#define DD       32
#define LL       5
#define OUTD     10
#define CATD     (LL * DD)         // 160
#define NTHREADS 256
#define NWAVES   8
#define STG_LD   40                // padded stride (halfs); rows stay 16B-aligned (80 B)

// ---- dynamic LDS layout (bytes); all sections 16B aligned ----
#define OFF_W0A   0                                  // f16 WT [32][128]
#define OFF_W0B   (OFF_W0A + 128*32*2)               // f16 WT [32][32]
#define OFF_W1    (OFF_W0B + 32*32*2)                // f16 WT [4][32][96]
#define OFF_W2    (OFF_W1  + 4*96*32*2)              // f16 WT [4][32][32]
#define OFF_BIAS  (OFF_W2  + 4*32*32*2)              // f32 [320]
#define OFF_EGO0  (OFF_BIAS + 320*4)                 // f16 [1024][32]
#define OFF_EGO1  (OFF_EGO0 + NPG_*DD*2)             // f16 [1024][32]
#define OFF_NEIG  (OFF_EGO1 + NPG_*DD*2)             // f32 [1024][32] (also pooled scratch)
#define OFF_STG   (OFF_NEIG + NPG_*DD*4)             // f16 [8][16][STG_LD]
#define OFF_COLOR (OFF_STG  + NWAVES*16*STG_LD*2)    // f32 [1024]
#define OFF_TOPK  (OFF_COLOR + NPG_*4)               // int [32]
#define OFF_BEST  (OFF_TOPK + 32*4)                  // u64
#define OFF_RED   (OFF_BEST + 16)                    // f32 [16]
#define SMEM_TOTAL (OFF_RED + 16*4)                  // ~320,976 B < 327,680 B

__device__ __forceinline__ v8f wmma32(v16h a, v16h b, v8f c) {
  return __builtin_amdgcn_wmma_f32_16x16x32_f16(false, a, false, b, (short)0, c,
                                                false, false);
}

__device__ __forceinline__ v16h combine16(v8h lo, v8h hi) {
  union { v16h v; v8h h[2]; } u;
  u.h[0] = lo; u.h[1] = hi;
  return u.v;
}

// B fragment from TRANSPOSED f16 weights WT[32][K]: lane col = (l&15)+nOff,
// K runs = kOff + (l>=16 ? 8 : 0) and +16, 8 halfs each -> 2x ds_load_b128.
__device__ __forceinline__ v16h load_b_fragT(const _Float16* WT, int K, int nOff, int kOff) {
  int l = threadIdx.x & 31;
  int col = (l & 15) + nOff;
  int kb = kOff + ((l >> 4) << 3);
  const v8h* p = (const v8h*)(WT + col * K + kb);
  return combine16(p[0], p[2]);
}

// A fragment (16x32) from global f32 rows; two contiguous 8-float runs per lane.
__device__ __forceinline__ v16h load_a_global_f32(const float* rowBase, int ld, int k0) {
  int l = threadIdx.x & 31;
  const float* r = rowBase + (size_t)(l & 15) * ld + k0 + ((l >> 4) << 3);
  f32x4 x0 = *(const f32x4*)(r);
  f32x4 x1 = *(const f32x4*)(r + 4);
  f32x4 x2 = *(const f32x4*)(r + 16);
  f32x4 x3 = *(const f32x4*)(r + 20);
  v16h a;
#pragma unroll
  for (int i = 0; i < 4; ++i) {
    a[i]      = (_Float16)x0[i];
    a[4 + i]  = (_Float16)x1[i];
    a[8 + i]  = (_Float16)x2[i];
    a[12 + i] = (_Float16)x3[i];
  }
  return a;
}

// A fragment from wave-private f16 stage tile [16][STG_LD]: 2x ds_load_b128.
__device__ __forceinline__ v16h load_a_stage(const _Float16* stg) {
  int l = threadIdx.x & 31;
  const _Float16* r = stg + (l & 15) * STG_LD + ((l >> 4) << 3);
  return combine16(*(const v8h*)r, *(const v8h*)(r + 16));
}

// A fragment for agg = [ego | neig | ego+neig]; each 32-wide K step is one mode.
__device__ __forceinline__ v16h load_a_agg(const _Float16* ego, const float* neig,
                                           int row0, int mode) {
  int l = threadIdx.x & 31;
  int row = row0 + (l & 15);
  int cb = (l >> 4) << 3;
  if (mode == 0) {
    const _Float16* er = ego + row * DD + cb;
    return combine16(*(const v8h*)er, *(const v8h*)(er + 16));
  }
  const float* nr = neig + row * DD + cb;
  f32x4 n0 = *(const f32x4*)(nr);
  f32x4 n1 = *(const f32x4*)(nr + 4);
  f32x4 n2 = *(const f32x4*)(nr + 16);
  f32x4 n3 = *(const f32x4*)(nr + 20);
  if (mode == 2) {
    const _Float16* er = ego + row * DD + cb;
    v8h e0 = *(const v8h*)er;
    v8h e1 = *(const v8h*)(er + 16);
#pragma unroll
    for (int i = 0; i < 4; ++i) {
      n0[i] += (float)e0[i];
      n1[i] += (float)e0[4 + i];
      n2[i] += (float)e1[i];
      n3[i] += (float)e1[4 + i];
    }
  }
  v16h a;
#pragma unroll
  for (int i = 0; i < 4; ++i) {
    a[i]      = (_Float16)n0[i];
    a[4 + i]  = (_Float16)n1[i];
    a[8 + i]  = (_Float16)n2[i];
    a[12 + i] = (_Float16)n3[i];
  }
  return a;
}

// hidden = relu(c + b1) -> f16 stage tile (D-matrix layout write)
__device__ __forceinline__ void bias_relu_to_stage(v8f c0, v8f c1, const float* bias,
                                                   _Float16* stg) {
  int l = threadIdx.x & 31;
  int n0 = l & 15, n1 = 16 + n0;
  int mb = (l >> 4) << 3;
  float bb0 = bias[n0], bb1 = bias[n1];
#pragma unroll
  for (int r = 0; r < 8; ++r) {
    int m = mb + r;
    stg[m * STG_LD + n0] = (_Float16)fmaxf(c0[r] + bb0, 0.f);
    stg[m * STG_LD + n1] = (_Float16)fmaxf(c1[r] + bb1, 0.f);
  }
}

// ego = d + b2 -> LDS ego buffer + global per-layer store; optional color capture (col 31)
__device__ __forceinline__ void store_ego_tile(v8f d0, v8f d1, const float* bias,
                                               _Float16* egoDst, int row0,
                                               _Float16* gOut, long gRowBase, int colBase,
                                               float* colorOut) {
  int l = threadIdx.x & 31;
  int n0 = l & 15, n1 = 16 + n0;
  int mb = row0 + ((l >> 4) << 3);
  float bb0 = bias[n0], bb1 = bias[n1];
#pragma unroll
  for (int r = 0; r < 8; ++r) {
    int m = mb + r;
    float v0 = d0[r] + bb0;
    float v1 = d1[r] + bb1;
    egoDst[m * DD + n0] = (_Float16)v0;
    egoDst[m * DD + n1] = (_Float16)v1;
    gOut[(gRowBase + m) * CATD + colBase + n0] = (_Float16)v0;
    gOut[(gRowBase + m) * CATD + colBase + n1] = (_Float16)v1;
    if (colorOut != nullptr && n1 == 31) colorOut[m] = v1;
  }
}

__global__ __launch_bounds__(NTHREADS, 1)
void ihgnn_graph_kernel(const float* __restrict__ nodeFeat,
                        const int* __restrict__ src, const int* __restrict__ dst,
                        const float* __restrict__ w0a, const float* __restrict__ b0a,
                        const float* __restrict__ w0b, const float* __restrict__ b0b,
                        const float* __restrict__ w1,  const float* __restrict__ mb1,
                        const float* __restrict__ w2,  const float* __restrict__ mb2,
                        const float* __restrict__ outW, const float* __restrict__ outB,
                        float* __restrict__ out, _Float16* __restrict__ egoStore) {
  extern __shared__ char smem[];
  const int tid  = threadIdx.x;
  const int wave = tid >> 5;
  const int lane = tid & 31;
  const int g    = blockIdx.x;
  const long g0  = (long)g * NPG_;

  _Float16* sW0aT = (_Float16*)(smem + OFF_W0A);   // [32][128]
  _Float16* sW0bT = (_Float16*)(smem + OFF_W0B);   // [32][32]
  _Float16* sW1T  = (_Float16*)(smem + OFF_W1);    // [4][32][96]
  _Float16* sW2T  = (_Float16*)(smem + OFF_W2);    // [4][32][32]
  float*    sBias = (float*)(smem + OFF_BIAS);     // [0:32) b0a [32:64) b0b [64:192) mb1 [192:320) mb2
  _Float16* sEgoA = (_Float16*)(smem + OFF_EGO0);
  _Float16* sEgoB = (_Float16*)(smem + OFF_EGO1);
  float*    sNeig = (float*)(smem + OFF_NEIG);
  _Float16* sStg  = (_Float16*)(smem + OFF_STG) + wave * 16 * STG_LD;
  float*    sColor = (float*)(smem + OFF_COLOR);
  int*      sTopk  = (int*)(smem + OFF_TOPK);
  unsigned long long* sBest = (unsigned long long*)(smem + OFF_BEST);
  float*    sRed   = (float*)(smem + OFF_RED);

  // ---- stage weights into LDS as TRANSPOSED f16 (WT[n][k]) ----
  for (int i = tid; i < FIN * DD; i += NTHREADS) {
    int k = i >> 5, n = i & 31;
    sW0aT[n * FIN + k] = (_Float16)w0a[i];
  }
  for (int i = tid; i < DD * DD; i += NTHREADS) {
    int k = i >> 5, n = i & 31;
    sW0bT[n * DD + k] = (_Float16)w0b[i];
  }
  for (int i = tid; i < 4 * 96 * DD; i += NTHREADS) {
    int l = i / (96 * DD), r = i - l * 96 * DD;
    int k = r >> 5, n = r & 31;
    sW1T[l * 32 * 96 + n * 96 + k] = (_Float16)w1[i];
  }
  for (int i = tid; i < 4 * DD * DD; i += NTHREADS) {
    int l = i / (DD * DD), r = i - l * DD * DD;
    int k = r >> 5, n = r & 31;
    sW2T[l * 32 * 32 + n * 32 + k] = (_Float16)w2[i];
  }
  if (tid < 32) { sBias[tid] = b0a[tid]; sBias[32 + tid] = b0b[tid]; }
  for (int i = tid; i < 128; i += NTHREADS) { sBias[64 + i] = mb1[i]; sBias[192 + i] = mb2[i]; }
  __syncthreads();

  // ---- layer 0: ego = relu(X @ W1 + b1) @ W2 + b2 ----
  {
    // hoist all B fragments into registers (reused by every row tile)
    v16h bA0[4], bA1[4];
#pragma unroll
    for (int ks = 0; ks < 4; ++ks) {
      bA0[ks] = load_b_fragT(sW0aT, FIN, 0, ks * 32);
      bA1[ks] = load_b_fragT(sW0aT, FIN, 16, ks * 32);
    }
    v16h bB0 = load_b_fragT(sW0bT, DD, 0, 0);
    v16h bB1 = load_b_fragT(sW0bT, DD, 16, 0);

    for (int t = wave; t < NPG_ / 16; t += NWAVES) {
      int row0 = t * 16;
      v8f c0 = {}, c1 = {};
      const float* aBase = nodeFeat + (g0 + row0) * FIN;
#pragma unroll
      for (int ks = 0; ks < 4; ++ks) {
        v16h a = load_a_global_f32(aBase, FIN, ks * 32);
        c0 = wmma32(a, bA0[ks], c0);
        c1 = wmma32(a, bA1[ks], c1);
      }
      bias_relu_to_stage(c0, c1, sBias, sStg);
      asm volatile("" ::: "memory");   // keep DS program order (HW DS ops in-order per wave)
      v16h a2 = load_a_stage(sStg);
      v8f d0 = {}, d1 = {};
      d0 = wmma32(a2, bB0, d0);
      d1 = wmma32(a2, bB1, d1);
      store_ego_tile(d0, d1, sBias + 32, sEgoA, row0, egoStore, g0, 0, nullptr);
    }
  }

  _Float16* egoCur = sEgoA;
  _Float16* egoNxt = sEgoB;

  // ---- layers 1..4: neighbor sum (LDS atomics) + MLP on [ego|neig|ego+neig] ----
  for (int l = 0; l < LL - 1; ++l) {
    __syncthreads();
    for (int i = tid; i < NPG_ * DD; i += NTHREADS) sNeig[i] = 0.f;
    __syncthreads();

    // scatter: lane = column -> conflict-free ds_add_f32 (32 consecutive dwords)
    {
      const long ebase = (long)g * EPG;
      for (int base = wave * 32; base < EPG; base += NWAVES * 32) {
        int myS = src[ebase + base + lane] - (int)g0;   // coalesced edge loads
        int myD = dst[ebase + base + lane] - (int)g0;
#pragma unroll 4
        for (int j = 0; j < 32; ++j) {
          int sl = __shfl(myS, j, 32);
          int dl = __shfl(myD, j, 32);
          float v = (float)egoCur[sl * DD + lane];
          atomicAdd(&sNeig[dl * DD + lane], v);         // ds_add_f32, banks 0..31
        }
      }
    }
    __syncthreads();

    const _Float16* W1lT = sW1T + l * 32 * 96;
    const _Float16* W2lT = sW2T + l * 32 * 32;
    const float* b1l = sBias + 64 + l * DD;
    const float* b2l = sBias + 192 + l * DD;
    float* colorDst = (l == LL - 2) ? sColor : nullptr;

    v16h bA0[3], bA1[3];
#pragma unroll
    for (int ks = 0; ks < 3; ++ks) {
      bA0[ks] = load_b_fragT(W1lT, 96, 0, ks * 32);
      bA1[ks] = load_b_fragT(W1lT, 96, 16, ks * 32);
    }
    v16h bB0 = load_b_fragT(W2lT, DD, 0, 0);
    v16h bB1 = load_b_fragT(W2lT, DD, 16, 0);

    for (int t = wave; t < NPG_ / 16; t += NWAVES) {
      int row0 = t * 16;
      v8f c0 = {}, c1 = {};
#pragma unroll
      for (int ks = 0; ks < 3; ++ks) {
        v16h a = load_a_agg(egoCur, sNeig, row0, ks);
        c0 = wmma32(a, bA0[ks], c0);
        c1 = wmma32(a, bA1[ks], c1);
      }
      bias_relu_to_stage(c0, c1, b1l, sStg);
      asm volatile("" ::: "memory");
      v16h a2 = load_a_stage(sStg);
      v8f d0 = {}, d1 = {};
      d0 = wmma32(a2, bB0, d0);
      d1 = wmma32(a2, bB1, d1);
      store_ego_tile(d0, d1, b2l, egoNxt, row0, egoStore, g0, (l + 1) * DD, colorDst);
    }
    _Float16* tmp = egoCur; egoCur = egoNxt; egoNxt = tmp;
  }
  __syncthreads();

  // ---- sort-pool: iterative top-30 on sColor (desc value, then smaller index) ----
  for (int k = 0; k < KTOP; ++k) {
    if (tid == 0) *sBest = 0ull;
    __syncthreads();
    unsigned long long best = 0ull;
    for (int n = tid; n < NPG_; n += NTHREADS) {
      unsigned ub = __float_as_uint(sColor[n]);
      ub = (ub & 0x80000000u) ? ~ub : (ub | 0x80000000u);   // orderable float
      unsigned long long key =
          ((unsigned long long)ub << 32) | (unsigned)(NPG_ - 1 - n);
      best = best > key ? best : key;
    }
    atomicMax(sBest, best);                                  // ds_max_u64
    __syncthreads();
    if (tid == 0) {
      int idx = NPG_ - 1 - (int)((unsigned)(*sBest & 0xFFFFFFFFull));
      sTopk[k] = idx;
      sColor[idx] = -__builtin_huge_valf();
    }
    __syncthreads();
  }

  // ---- gather pooled [30][160] into LDS (reuse sNeig region), final dense + relu ----
  float* pooled = sNeig;
  for (int i = tid; i < KTOP * CATD; i += NTHREADS) {
    int k = i / CATD, c = i - k * CATD;
    long node = g0 + sTopk[k];
    pooled[i] = (float)egoStore[node * CATD + c];
  }
  __syncthreads();

  float part[OUTD];
#pragma unroll
  for (int o = 0; o < OUTD; ++o) part[o] = 0.f;
  for (int i = tid; i < KTOP * CATD; i += NTHREADS) {
    float x = pooled[i];
    const float* w = outW + i * OUTD;
#pragma unroll
    for (int o = 0; o < OUTD; ++o) part[o] += x * w[o];
  }
  if (tid < OUTD) sRed[tid] = 0.f;
  __syncthreads();
#pragma unroll
  for (int o = 0; o < OUTD; ++o) atomicAdd(&sRed[o], part[o]);
  __syncthreads();
  if (tid < OUTD) out[g * OUTD + tid] = fmaxf(sRed[tid] + outB[tid], 0.f);
}

extern "C" void kernel_launch(void* const* d_in, const int* in_sizes, int n_in,
                              void* d_out, int out_size, void* d_ws, size_t ws_size,
                              hipStream_t stream) {
  (void)in_sizes; (void)n_in; (void)out_size; (void)ws_size;
  const float* nodeFeat = (const float*)d_in[0];
  const int*   src      = (const int*)d_in[1];
  const int*   dst      = (const int*)d_in[2];
  const float* w0a      = (const float*)d_in[3];
  const float* b0a      = (const float*)d_in[4];
  const float* w0b      = (const float*)d_in[5];
  const float* b0b      = (const float*)d_in[6];
  const float* w1       = (const float*)d_in[7];
  const float* mb1      = (const float*)d_in[8];
  const float* w2       = (const float*)d_in[9];
  const float* mb2      = (const float*)d_in[10];
  const float* outW     = (const float*)d_in[11];
  const float* outB     = (const float*)d_in[12];
  float* out = (float*)d_out;
  _Float16* egoStore = (_Float16*)d_ws;   // [N][160] f16 per-layer ego concat (~160 MB)

  ihgnn_graph_kernel<<<G_, NTHREADS, SMEM_TOTAL, stream>>>(
      nodeFeat, src, dst, w0a, b0a, w0b, b0b, w1, mb1, w2, mb2, outW, outB,
      out, egoStore);
}
